// HomoGATEncoderLinearDropoutHiddenChannels_15805479649923
// MI455X (gfx1250) — matrix operand order repaired and checked by reading, not compiled, not run on hardware
//
#include <hip/hip_runtime.h>
#include <hip/hip_fp16.h>

typedef __attribute__((ext_vector_type(16))) _Float16 v16h;
typedef __attribute__((ext_vector_type(8)))  float    v8f;

#define NEG_SLOPE 0.2f

// ---------------------------------------------------------------- utilities
__global__ void k_zero_u32(unsigned int* __restrict__ p, long long n) {
    long long i = (long long)blockIdx.x * blockDim.x + threadIdx.x;
    if (i < n) p[i] = 0u;
}

__global__ void k_f32_to_f16(const float* __restrict__ s, _Float16* __restrict__ d, int n) {
    int i = blockIdx.x * blockDim.x + threadIdx.x;
    if (i < n) d[i] = (_Float16)s[i];
}

// weight convert + transpose: s is [K,Nc] row-major f32, d becomes [Nc,K] f16
__global__ void k_w_to_f16_t(const float* __restrict__ s, _Float16* __restrict__ d,
                             int K, int Nc) {
    int i = blockIdx.x * blockDim.x + threadIdx.x;
    if (i >= K * Nc) return;
    int k = i / Nc, n = i - k * Nc;
    d[(size_t)n * K + k] = (_Float16)s[i];
}

__device__ __forceinline__ float elu(float v) { return v > 0.f ? v : (__expf(v) - 1.f); }

// ------------------------------------------------- WMMA GEMM: C = A@B + bias
// A:  [M,K] f16 row-major.  Bt: [Nc,K] f16 (B transposed -> contiguous lane frags).
// One wave computes a 16x(16*NPW) strip; A fragments cached in registers.
// K, Nc, ACT are compile-time so all strides/offsets are immediates.
template <int K, int Nc, bool ACT>
__global__ void k_gemm_wmma_f16(const _Float16* __restrict__ A,
                                const _Float16* __restrict__ Bt,
                                const float* __restrict__ bias,
                                float* __restrict__ C,
                                int M, int MT, int NTG) {
    constexpr int NPW = 4;
    constexpr int NK  = K / 32;
    constexpr int NT  = Nc / 16;
    int wave = blockIdx.x * (blockDim.x >> 5) + (threadIdx.x >> 5);
    if (wave >= MT * NTG) return;           // wave-uniform exit: EXEC stays all-1 for WMMA
    int ntg = wave % NTG;
    int mt  = wave / NTG;
    int lane = threadIdx.x & 31;
    int l15  = lane & 15;
    int kh   = lane >> 4;

    int arow = mt * 16 + l15;               // A row (both lane halves hold same rows)
    if (arow >= M) arow = M - 1;            // clamp; stores guarded below
    const _Float16* Abase = A + (size_t)arow * K + kh * 8;

    v16h afrag[NK];
#pragma unroll
    for (int ki = 0; ki < NK; ++ki) {
        const _Float16* ap = Abase + ki * 32;   // two contiguous 16B chunks
#pragma unroll
        for (int j = 0; j < 8; ++j) { afrag[ki][j] = ap[j]; afrag[ki][j + 8] = ap[16 + j]; }
    }

    const bool fullTile = (mt * 16 + 16) <= M;   // wave-uniform
#pragma unroll
    for (int t = 0; t < NPW; ++t) {
        int nt = ntg * NPW + t;
        if (NT > NPW ? (nt >= NT) : (t >= NT)) break;   // uniform; folds when NT<=NPW
        int col = nt * 16 + l15;
        float bv = bias[col];
        v8f acc;
#pragma unroll
        for (int r = 0; r < 8; ++r) acc[r] = bv;

        const _Float16* bp = Bt + (size_t)col * K + kh * 8;   // two 8-elem chunks per lane
#pragma unroll
        for (int ki = 0; ki < NK; ++ki) {
            v16h b;
#pragma unroll
            for (int j = 0; j < 8; ++j) { b[j] = bp[ki * 32 + j]; b[j + 8] = bp[ki * 32 + 16 + j]; }  // 2x b128
            acc = __builtin_amdgcn_wmma_f32_16x16x32_f16(
                false, afrag[ki], false, b, (short)0, acc, false, false);
        }

        // D layout: lane half kh holds rows kh*8 + r
        float* cp = C + (size_t)(mt * 16 + kh * 8) * Nc + col;
        if (fullTile) {                       // fast path: 8 stores w/ immediate offsets
#pragma unroll
            for (int r = 0; r < 8; ++r) {
                float v = acc[r];
                cp[r * Nc] = ACT ? elu(v) : v;
            }
        } else {                              // partial last M-tile only
#pragma unroll
            for (int r = 0; r < 8; ++r) {
                if (mt * 16 + kh * 8 + r < M) {
                    float v = acc[r];
                    cp[r * Nc] = ACT ? elu(v) : v;
                }
            }
        }
    }
}

// ----------------------------------------------------------- edge machinery
__device__ __forceinline__ void edge_sd(const int* __restrict__ ei, int E, int e,
                                        int& s, int& d) {
    if (e < E) { s = ei[e]; d = ei[E + e]; }
    else       { s = d = e - E; }           // appended self-loops
}

__device__ __forceinline__ unsigned int fenc(float f) {   // monotone float->u32
    unsigned int u = __float_as_uint(f);
    return (u & 0x80000000u) ? ~u : (u | 0x80000000u);
}
__device__ __forceinline__ float fdec(unsigned int u) {
    return (u & 0x80000000u) ? __uint_as_float(u & 0x7fffffffu) : __uint_as_float(~u);
}
__device__ __forceinline__ float lrelu(float v) { return v > 0.f ? v : NEG_SLOPE * v; }

// pass 1: logits[e,h] = att_h . leaky_relu(xl[s,h,:] + xr[d,h,:]); segment max via atomicMax
__global__ void k_edge_logits(const int* __restrict__ ei,
                              const float* __restrict__ xl, const float* __restrict__ xr,
                              const float* __restrict__ att,
                              float* __restrict__ logits, unsigned int* __restrict__ menc,
                              int E, int N, int H, int Cc) {
    int tid = blockIdx.x * blockDim.x + threadIdx.x;
    if (tid >= (E + N) * H) return;
    int e = tid / H, h = tid - e * H;
    int s, d; edge_sd(ei, E, e, s, d);
    const float4* pl = (const float4*)(xl + ((size_t)s * H + h) * Cc);
    const float4* pr = (const float4*)(xr + ((size_t)d * H + h) * Cc);
    const float4* pa = (const float4*)(att + h * Cc);
    float acc = 0.f;
    for (int c = 0; c < (Cc >> 2); ++c) {
        float4 a = pl[c], b = pr[c], w = pa[c];
        acc += w.x * lrelu(a.x + b.x) + w.y * lrelu(a.y + b.y)
             + w.z * lrelu(a.z + b.z) + w.w * lrelu(a.w + b.w);
    }
    logits[(size_t)e * H + h] = acc;
    atomicMax(&menc[(size_t)d * H + h], fenc(acc));
}

// pass 2: ex = exp(logit - max[d]); denom[d] += ex  (ex overwrites logits in place)
__global__ void k_edge_exp(const int* __restrict__ ei,
                           float* __restrict__ logits,
                           const unsigned int* __restrict__ menc,
                           float* __restrict__ denom, int E, int N, int H) {
    int tid = blockIdx.x * blockDim.x + threadIdx.x;
    if (tid >= (E + N) * H) return;
    int e = tid / H, h = tid - e * H;
    int s, d; edge_sd(ei, E, e, s, d);
    float m  = fdec(menc[(size_t)d * H + h]);
    float ex = __expf(logits[(size_t)e * H + h] - m);
    logits[(size_t)e * H + h] = ex;
    atomicAdd(&denom[(size_t)d * H + h], ex);
}

// pass 3: out[d,h,:] += xl[s,h,:] * (ex / denom[d,h])
__global__ void k_edge_aggr(const int* __restrict__ ei,
                            const float* __restrict__ xl,
                            const float* __restrict__ ex,
                            const float* __restrict__ denom,
                            float* __restrict__ out, int E, int N, int H, int Cc) {
    int tid = blockIdx.x * blockDim.x + threadIdx.x;
    if (tid >= (E + N) * H) return;
    int e = tid / H, h = tid - e * H;
    int s, d; edge_sd(ei, E, e, s, d);
    float alpha = ex[(size_t)e * H + h] / denom[(size_t)d * H + h];
    const float4* pl = (const float4*)(xl + ((size_t)s * H + h) * Cc);
    float* po = out + ((size_t)d * H + h) * Cc;
    for (int c = 0; c < (Cc >> 2); ++c) {
        float4 a = pl[c];
        atomicAdd(&po[4 * c + 0], a.x * alpha);
        atomicAdd(&po[4 * c + 1], a.y * alpha);
        atomicAdd(&po[4 * c + 2], a.z * alpha);
        atomicAdd(&po[4 * c + 3], a.w * alpha);
    }
}

// pass 4: h = elu(out + bias), stored f16 for the next WMMA GEMM
__global__ void k_finish_elu_f16(const float* __restrict__ out, const float* __restrict__ bias,
                                 _Float16* __restrict__ hf, int N, int W) {
    int i = blockIdx.x * blockDim.x + threadIdx.x;
    if (i >= N * W) return;
    float v = out[i] + bias[i % W];
    hf[i] = (_Float16)elu(v);
}

// ------------------------------------------------------------------- driver
static inline void* wtake(char* base, size_t& off, size_t bytes) {
    void* p = base + off;
    off += (bytes + 255) & ~(size_t)255;
    return p;
}

extern "C" void kernel_launch(void* const* d_in, const int* in_sizes, int n_in,
                              void* d_out, int out_size, void* d_ws, size_t ws_size,
                              hipStream_t stream) {
    const float* x     = (const float*)d_in[0];
    const int*   ei    = (const int*)d_in[1];
    const float* Wl1   = (const float*)d_in[2];
    const float* bl1   = (const float*)d_in[3];
    const float* Wr1   = (const float*)d_in[4];
    const float* br1   = (const float*)d_in[5];
    const float* att1  = (const float*)d_in[6];
    const float* bias1 = (const float*)d_in[7];
    const float* Wl2   = (const float*)d_in[8];
    const float* bl2   = (const float*)d_in[9];
    const float* Wr2   = (const float*)d_in[10];
    const float* br2   = (const float*)d_in[11];
    const float* att2  = (const float*)d_in[12];
    const float* bias2 = (const float*)d_in[13];
    const float* Wlin  = (const float*)d_in[14];
    const float* blin  = (const float*)d_in[15];

    const int F = 128, H1 = 8, C1 = 16, HC1 = H1 * C1, C2 = 64, OUT = 128;
    const int N = in_sizes[0] / F;
    const int E = in_sizes[1] / 2;
    const int Et = E + N;

    // workspace carving (buffers reused across layers).
    // MX, DN, O1 are carved contiguously so one fused zero launch covers all three.
    size_t off = 0; char* ws = (char*)d_ws;
    _Float16*     Af16 = (_Float16*)    wtake(ws, off, (size_t)N * 128 * 2); // x / h1 / h2 f16
    _Float16*     Wf16 = (_Float16*)    wtake(ws, off, (size_t)128 * 128 * 2); // transposed wt
    float*        XL   = (float*)       wtake(ws, off, (size_t)N * 128 * 4);
    float*        XR   = (float*)       wtake(ws, off, (size_t)N * 128 * 4);
    float*        LG   = (float*)       wtake(ws, off, (size_t)Et * 8 * 4);  // logits -> ex
    unsigned int* MX   = (unsigned int*)wtake(ws, off, (size_t)N * 8 * 4);
    float*        DN   = (float*)       wtake(ws, off, (size_t)N * 8 * 4);
    float*        O1   = (float*)       wtake(ws, off, (size_t)N * 128 * 4); // aggregated out
    char*         zend = ws + off;
    (void)ws_size; (void)n_in; (void)out_size;

    const int TB = 256;
    auto gb = [](long long n, int tb) { return (int)((n + tb - 1) / tb); };

    // fused zero of MX + DN + O1 (contiguous span, padding included)
    long long nzero = (long long)(zend - (char*)MX) / 4;
    auto zero_accum = [&]() {
        hipLaunchKernelGGL(k_zero_u32, dim3(gb(nzero, TB)), dim3(TB), 0, stream, MX, nzero);
    };

    // GEMM helper: one wave per 16x64 output strip (up to 4 n-tiles), 8 waves per block
    auto ggrid = [&](int M, int Nc, int& MT, int& NTG) {
        MT = (M + 15) / 16; int NT = Nc / 16; NTG = (NT + 3) / 4;
        return dim3((MT * NTG + 7) / 8);
    };
    int MT, NTG;

    // ================= layer 1 (H=8, C=16) =================
    hipLaunchKernelGGL(k_f32_to_f16, dim3(gb((long long)N * F, TB)), dim3(TB), 0, stream,
                       x, Af16, N * F);
    hipLaunchKernelGGL(k_w_to_f16_t, dim3(gb(F * HC1, TB)), dim3(TB), 0, stream, Wl1, Wf16, F, HC1);
    hipLaunchKernelGGL((k_gemm_wmma_f16<128, 128, false>), ggrid(N, HC1, MT, NTG), dim3(256), 0,
                       stream, Af16, Wf16, bl1, XL, N, MT, NTG);
    hipLaunchKernelGGL(k_w_to_f16_t, dim3(gb(F * HC1, TB)), dim3(TB), 0, stream, Wr1, Wf16, F, HC1);
    hipLaunchKernelGGL((k_gemm_wmma_f16<128, 128, false>), ggrid(N, HC1, MT, NTG), dim3(256), 0,
                       stream, Af16, Wf16, br1, XR, N, MT, NTG);

    zero_accum();
    hipLaunchKernelGGL(k_edge_logits, dim3(gb((long long)Et * H1, TB)), dim3(TB), 0, stream,
                       ei, XL, XR, att1, LG, MX, E, N, H1, C1);
    hipLaunchKernelGGL(k_edge_exp, dim3(gb((long long)Et * H1, TB)), dim3(TB), 0, stream,
                       ei, LG, MX, DN, E, N, H1);
    hipLaunchKernelGGL(k_edge_aggr, dim3(gb((long long)Et * H1, TB)), dim3(TB), 0, stream,
                       ei, XL, LG, DN, O1, E, N, H1, C1);
    hipLaunchKernelGGL(k_finish_elu_f16, dim3(gb((long long)N * HC1, TB)), dim3(TB), 0, stream,
                       O1, bias1, Af16, N, HC1);   // h1 (f16) into Af16

    // ================= layer 2 (H=1, C=64) =================
    hipLaunchKernelGGL(k_w_to_f16_t, dim3(gb(HC1 * C2, TB)), dim3(TB), 0, stream, Wl2, Wf16, HC1, C2);
    hipLaunchKernelGGL((k_gemm_wmma_f16<128, 64, false>), ggrid(N, C2, MT, NTG), dim3(256), 0,
                       stream, Af16, Wf16, bl2, XL, N, MT, NTG);
    hipLaunchKernelGGL(k_w_to_f16_t, dim3(gb(HC1 * C2, TB)), dim3(TB), 0, stream, Wr2, Wf16, HC1, C2);
    hipLaunchKernelGGL((k_gemm_wmma_f16<128, 64, false>), ggrid(N, C2, MT, NTG), dim3(256), 0,
                       stream, Af16, Wf16, br2, XR, N, MT, NTG);

    zero_accum();
    hipLaunchKernelGGL(k_edge_logits, dim3(gb((long long)Et, TB)), dim3(TB), 0, stream,
                       ei, XL, XR, att2, LG, MX, E, N, 1, C2);
    hipLaunchKernelGGL(k_edge_exp, dim3(gb((long long)Et, TB)), dim3(TB), 0, stream,
                       ei, LG, MX, DN, E, N, 1);
    hipLaunchKernelGGL(k_edge_aggr, dim3(gb((long long)Et, TB)), dim3(TB), 0, stream,
                       ei, XL, LG, DN, O1, E, N, 1, C2);
    hipLaunchKernelGGL(k_finish_elu_f16, dim3(gb((long long)N * C2, TB)), dim3(TB), 0, stream,
                       O1, bias2, Af16, N, C2);    // h2 (f16) into Af16

    // ================= final linear + ELU ==================
    hipLaunchKernelGGL(k_w_to_f16_t, dim3(gb(C2 * OUT, TB)), dim3(TB), 0, stream, Wlin, Wf16, C2, OUT);
    hipLaunchKernelGGL((k_gemm_wmma_f16<64, 128, true>), ggrid(N, OUT, MT, NTG), dim3(256), 0,
                       stream, Af16, Wf16, blin, (float*)d_out, N, MT, NTG);
}
